// EncoderBlock_53077205844365
// MI455X (gfx1250) — matrix operand order repaired and checked by reading, not compiled, not run on hardware
//
#include <hip/hip_runtime.h>
#include <hip/hip_bf16.h>

typedef __attribute__((ext_vector_type(16))) __bf16 v16bf;
typedef __attribute__((ext_vector_type(8)))  float  v8f;
typedef unsigned short ushort_t;

#define D_MODEL 1024
#define SEQ     2048
#define NHEADS  16
#define HD      64
#define D_FF    4096
#define M_TOT   (2 * SEQ)   // 4096 rows

// ---------- helpers ----------

static __device__ __forceinline__ ushort_t f2bf(float f) {
    unsigned int u = __float_as_uint(f);
    unsigned int r = (u + 0x7FFFu + ((u >> 16) & 1u)) >> 16;
    return (ushort_t)r;
}

union FragU { v16bf v; uint4 q[2]; ushort_t s[16]; };

// Generic (flat) shared pointer -> 32-bit LDS byte offset (flat aperture: low 32 bits).
static __device__ __forceinline__ unsigned lds_addr32(const void* p) {
    return (unsigned)(unsigned long long)(const __attribute__((address_space(3))) char*)p;
}

// Async 16B copy global -> LDS (CDNA5 GLOBAL_LOAD_ASYNC_TO_LDS_B128, ASYNCcnt).
static __device__ __forceinline__ void async_cp_b128(void* lds_dst, const void* gsrc) {
    unsigned l = lds_addr32(lds_dst);
    unsigned long long g = (unsigned long long)gsrc;
    asm volatile("global_load_async_to_lds_b128 %0, %1, off"
                 :: "v"(l), "v"(g) : "memory");
}

static __device__ __forceinline__ void wait_async() {
    asm volatile("s_wait_asynccnt 0x0" ::: "memory");
}

// A-fragment: 16x32 bf16 tile, row-major in LDS with leading dim ldm (elements).
// Lanes 0-15 (g=0): K=0..15 ; lanes 16-31 (g=1): K=16..31 (matches B fragments).
// Contiguous 16B per half-frag.
static __device__ __forceinline__ v16bf load_frag_a(const ushort_t* base, int ldm, int lane) {
    int g = (lane >> 4) & 1;
    int m = lane & 15;
    FragU u;
    u.q[0] = *(const uint4*)(base + m * ldm + 16 * g);
    u.q[1] = *(const uint4*)(base + m * ldm + 16 * g + 8);
    return u.v;
}

// B-fragment from Bt (N x K row-major tile in LDS, leading dim ldk).
// Lane group g=0 covers K=0..15, g=1 covers K=16..31; col N = lane&15.
static __device__ __forceinline__ v16bf load_frag_bt(const ushort_t* base, int ldk, int lane) {
    int g = (lane >> 4) & 1;
    int n = lane & 15;
    FragU u;
    u.q[0] = *(const uint4*)(base + n * ldk + 16 * g);
    u.q[1] = *(const uint4*)(base + n * ldk + 16 * g + 8);
    return u.v;
}

// B-fragment from a row-major K x N tile (leading dim ldn) -- used for P x V.
static __device__ __forceinline__ v16bf load_frag_b_rm(const ushort_t* base, int ldn, int lane) {
    int g = (lane >> 4) & 1;
    int n = lane & 15;
    FragU u;
#pragma unroll
    for (int j = 0; j < 8; ++j) {
        int k = 16 * g + 2 * j;
        u.s[2 * j]     = base[k * ldn + n];
        u.s[2 * j + 1] = base[(k + 1) * ldn + n];
    }
    return u.v;
}

static __device__ __forceinline__ float hmax16(float v) {
    v = fmaxf(v, __shfl_xor(v, 1, 32));
    v = fmaxf(v, __shfl_xor(v, 2, 32));
    v = fmaxf(v, __shfl_xor(v, 4, 32));
    v = fmaxf(v, __shfl_xor(v, 8, 32));
    return v;
}
static __device__ __forceinline__ float hsum16(float v) {
    v += __shfl_xor(v, 1, 32);
    v += __shfl_xor(v, 2, 32);
    v += __shfl_xor(v, 4, 32);
    v += __shfl_xor(v, 8, 32);
    return v;
}

// ---------- conversion kernels ----------

__global__ void cvt_f32_bf16(const float* __restrict__ in, ushort_t* __restrict__ out, int n) {
    int i = blockIdx.x * 256 + threadIdx.x;
    if (i < n) out[i] = f2bf(in[i]);
}

// W: K x N fp32  ->  Wt: N x K bf16   (coalesced reads, strided writes)
__global__ void transpose_cvt(const float* __restrict__ W, ushort_t* __restrict__ Wt, int K, int N) {
    long long idx = (long long)blockIdx.x * 256 + threadIdx.x;
    if (idx >= (long long)K * N) return;
    int k = (int)(idx / N);
    int n = (int)(idx % N);
    Wt[(long long)n * K + k] = f2bf(W[idx]);
}

// ---------- GEMM: C(MxN) = A(MxK, bf16) * W(KxN) + bias ; Bt = N x K bf16 ----------

__global__ __launch_bounds__(256)
void gemm_bf16(const ushort_t* __restrict__ A, const ushort_t* __restrict__ Bt,
               const float* __restrict__ bias, float* __restrict__ Cf,
               ushort_t* __restrict__ Cb, int M, int N, int K, int relu) {
    __shared__ ushort_t As[2][128 * 32];
    __shared__ ushort_t Bs[2][128 * 32];

    const int tid  = threadIdx.x;
    const int wave = tid >> 5;
    const int lane = tid & 31;
    const int g    = (lane >> 4) & 1;
    const int ln   = lane & 15;
    const int n0   = blockIdx.x * 128;
    const int m0   = blockIdx.y * 128;

    // per-thread staging coordinates (2 x b128 for A, 2 x b128 for B per chunk)
    const int i0r = tid >> 2, i0c = (tid & 3) * 8;
    const int i1r = (tid + 256) >> 2, i1c = ((tid + 256) & 3) * 8;

    v8f acc[8] = {};

    // prologue: stage first K-chunk into buffer 0
    {
        async_cp_b128(&As[0][i0r * 32 + i0c], &A[(size_t)(m0 + i0r) * K + i0c]);
        async_cp_b128(&As[0][i1r * 32 + i1c], &A[(size_t)(m0 + i1r) * K + i1c]);
        async_cp_b128(&Bs[0][i0r * 32 + i0c], &Bt[(size_t)(n0 + i0r) * K + i0c]);
        async_cp_b128(&Bs[0][i1r * 32 + i1c], &Bt[(size_t)(n0 + i1r) * K + i1c]);
    }

    int cur = 0;
    for (int k0 = 0; k0 < K; k0 += 32) {
        wait_async();
        __syncthreads();   // buffer[cur] fully written; previous reads done

        if (k0 + 32 < K) {
            int nb = cur ^ 1, kn = k0 + 32;
            async_cp_b128(&As[nb][i0r * 32 + i0c], &A[(size_t)(m0 + i0r) * K + kn + i0c]);
            async_cp_b128(&As[nb][i1r * 32 + i1c], &A[(size_t)(m0 + i1r) * K + kn + i1c]);
            async_cp_b128(&Bs[nb][i0r * 32 + i0c], &Bt[(size_t)(n0 + i0r) * K + kn + i0c]);
            async_cp_b128(&Bs[nb][i1r * 32 + i1c], &Bt[(size_t)(n0 + i1r) * K + kn + i1c]);
        }

        v16bf a = load_frag_a(&As[cur][(16 * wave) * 32], 32, lane);
        v16bf bf[8];
#pragma unroll
        for (int sub = 0; sub < 8; ++sub)
            bf[sub] = load_frag_bt(&Bs[cur][(16 * sub) * 32], 32, lane);
#pragma unroll
        for (int sub = 0; sub < 8; ++sub)
            acc[sub] = __builtin_amdgcn_wmma_f32_16x16x32_bf16(
                false, a, false, bf[sub], (short)0, acc[sub], false, false);

        cur ^= 1;
    }

#pragma unroll
    for (int sub = 0; sub < 8; ++sub) {
        int col  = n0 + 16 * sub + ln;
        float bv = bias ? bias[col] : 0.0f;
#pragma unroll
        for (int v = 0; v < 8; ++v) {
            int row   = m0 + 16 * wave + v + 8 * g;
            float val = acc[sub][v] + bv;
            if (relu) val = fmaxf(val, 0.0f);
            if (Cf) Cf[(size_t)row * N + col] = val;
            if (Cb) Cb[(size_t)row * N + col] = f2bf(val);
        }
    }
}

// ---------- flash attention: per (b,h), 64-query tile ----------

__global__ __launch_bounds__(128)
void attn_kernel(const ushort_t* __restrict__ Qb, const ushort_t* __restrict__ Kb,
                 const ushort_t* __restrict__ Vb, ushort_t* __restrict__ Cb) {
    __shared__ ushort_t Qs[64 * 64];
    __shared__ ushort_t Ks[64 * 64];
    __shared__ ushort_t Vs[64 * 64];
    __shared__ ushort_t Ps[4][16 * 64];

    const int tid  = threadIdx.x;
    const int wave = tid >> 5;
    const int lane = tid & 31;
    const int g    = (lane >> 4) & 1;
    const int ln   = lane & 15;
    const int q0   = blockIdx.x * 64;
    const int bh   = blockIdx.y;
    const int b    = bh >> 4;
    const int h    = bh & 15;
    const size_t rowbase = (size_t)b * SEQ;
    const int coloff = h * HD;

    // async-stage Q tile (4 x b128 per thread)
#pragma unroll
    for (int r = 0; r < 4; ++r) {
        int idx = tid + 128 * r;
        int row = idx >> 3, c4 = (idx & 7) * 8;
        async_cp_b128(&Qs[row * 64 + c4],
                      &Qb[(rowbase + q0 + row) * D_MODEL + coloff + c4]);
    }

    v8f acc_o[4] = {};
    float mrow[8], lrow[8];
#pragma unroll
    for (int v = 0; v < 8; ++v) { mrow[v] = -3.0e38f; lrow[v] = 0.0f; }

    for (int j0 = 0; j0 < SEQ; j0 += 64) {
        __syncthreads();   // everyone done reading previous K/V
#pragma unroll
        for (int r = 0; r < 4; ++r) {
            int idx = tid + 128 * r;
            int row = idx >> 3, c4 = (idx & 7) * 8;
            async_cp_b128(&Ks[row * 64 + c4],
                          &Kb[(rowbase + j0 + row) * D_MODEL + coloff + c4]);
            async_cp_b128(&Vs[row * 64 + c4],
                          &Vb[(rowbase + j0 + row) * D_MODEL + coloff + c4]);
        }
        wait_async();      // also covers the Q staging on the first iteration
        __syncthreads();

        // scores = Q * K^T  (Bt tile == K tile, n-major)
        v8f sc[4] = {};
#pragma unroll
        for (int kc = 0; kc < 64; kc += 32) {
            v16bf aq = load_frag_a(&Qs[(16 * wave) * 64 + kc], 64, lane);
#pragma unroll
            for (int sub = 0; sub < 4; ++sub) {
                v16bf bk = load_frag_bt(&Ks[(16 * sub) * 64 + kc], 64, lane);
                sc[sub] = __builtin_amdgcn_wmma_f32_16x16x32_bf16(
                    false, aq, false, bk, (short)0, sc[sub], false, false);
            }
        }

        const float scale = 0.125f;  // 1/sqrt(64)
#pragma unroll
        for (int v = 0; v < 8; ++v) {
            float mx = -3.0e38f;
#pragma unroll
            for (int sub = 0; sub < 4; ++sub) mx = fmaxf(mx, sc[sub][v] * scale);
            mx = hmax16(mx);
            float mnew  = fmaxf(mrow[v], mx);
            float alpha = __expf(mrow[v] - mnew);
            float ps = 0.0f;
#pragma unroll
            for (int sub = 0; sub < 4; ++sub) {
                float p = __expf(sc[sub][v] * scale - mnew);
                ps += p;
                Ps[wave][(v + 8 * g) * 64 + 16 * sub + ln] = f2bf(p);
            }
            ps = hsum16(ps);
            lrow[v] = lrow[v] * alpha + ps;
            mrow[v] = mnew;
#pragma unroll
            for (int sub = 0; sub < 4; ++sub) acc_o[sub][v] *= alpha;
        }

        // ctx += P * V   (V row-major K x N tile)
#pragma unroll
        for (int kc = 0; kc < 64; kc += 32) {
            v16bf ap = load_frag_a(&Ps[wave][kc], 64, lane);
#pragma unroll
            for (int sub = 0; sub < 4; ++sub) {
                v16bf bv = load_frag_b_rm(&Vs[kc * 64 + 16 * sub], 64, lane);
                acc_o[sub] = __builtin_amdgcn_wmma_f32_16x16x32_bf16(
                    false, ap, false, bv, (short)0, acc_o[sub], false, false);
            }
        }
    }

#pragma unroll
    for (int v = 0; v < 8; ++v) {
        float inv = 1.0f / lrow[v];
        int m = v + 8 * g;
        size_t row = rowbase + q0 + 16 * wave + m;
#pragma unroll
        for (int sub = 0; sub < 4; ++sub)
            Cb[row * D_MODEL + coloff + 16 * sub + ln] = f2bf(acc_o[sub][v] * inv);
    }
}

// ---------- fused residual + LayerNorm ----------

__global__ __launch_bounds__(256)
void add_ln_kernel(const float* __restrict__ x, const float* __restrict__ y,
                   const float* __restrict__ gam, const float* __restrict__ bet,
                   float* __restrict__ outf, ushort_t* __restrict__ outb) {
    __shared__ float r1[256];
    __shared__ float r2[256];
    const int row = blockIdx.x;
    const int tid = threadIdx.x;
    const float* xr = x + (size_t)row * D_MODEL;
    const float* yr = y + (size_t)row * D_MODEL;

    float v[4];
    float s = 0.0f, s2 = 0.0f;
#pragma unroll
    for (int i = 0; i < 4; ++i) {
        int c = tid + 256 * i;
        v[i] = xr[c] + yr[c];
        s += v[i];
        s2 += v[i] * v[i];
    }
    r1[tid] = s; r2[tid] = s2;
    __syncthreads();
    for (int st = 128; st > 0; st >>= 1) {
        if (tid < st) { r1[tid] += r1[tid + st]; r2[tid] += r2[tid + st]; }
        __syncthreads();
    }
    float mean = r1[0] * (1.0f / D_MODEL);
    float var  = r2[0] * (1.0f / D_MODEL) - mean * mean;
    float rstd = rsqrtf(var + 1e-5f);
#pragma unroll
    for (int i = 0; i < 4; ++i) {
        int c = tid + 256 * i;
        float o = (v[i] - mean) * rstd * gam[c] + bet[c];
        if (outf) outf[(size_t)row * D_MODEL + c] = o;
        if (outb) outb[(size_t)row * D_MODEL + c] = f2bf(o);
    }
}

// ---------- launcher ----------

extern "C" void kernel_launch(void* const* d_in, const int* in_sizes, int n_in,
                              void* d_out, int out_size, void* d_ws, size_t ws_size,
                              hipStream_t stream) {
    const float* x     = (const float*)d_in[0];
    const float* w_q   = (const float*)d_in[1];
    const float* b_q   = (const float*)d_in[2];
    const float* w_k   = (const float*)d_in[3];
    const float* b_k   = (const float*)d_in[4];
    const float* w_v   = (const float*)d_in[5];
    const float* b_v   = (const float*)d_in[6];
    const float* w_o   = (const float*)d_in[7];
    const float* b_o   = (const float*)d_in[8];
    const float* w_ff1 = (const float*)d_in[9];
    const float* b_ff1 = (const float*)d_in[10];
    const float* w_ff2 = (const float*)d_in[11];
    const float* b_ff2 = (const float*)d_in[12];
    const float* ln1_g = (const float*)d_in[13];
    const float* ln1_b = (const float*)d_in[14];
    const float* ln2_g = (const float*)d_in[15];
    const float* ln2_b = (const float*)d_in[16];
    float* out = (float*)d_out;

    char* ws = (char*)d_ws;
    size_t off = 0;
    auto alloc = [&](size_t bytes) -> char* {
        char* p = ws + off;
        off += (bytes + 255) & ~(size_t)255;
        return p;
    };
    const size_t MD2 = (size_t)M_TOT * D_MODEL * 2;
    const size_t MD4 = (size_t)M_TOT * D_MODEL * 4;
    const size_t MF2 = (size_t)M_TOT * D_FF * 2;
    const size_t DD2 = (size_t)D_MODEL * D_MODEL * 2;
    const size_t DF2 = (size_t)D_MODEL * D_FF * 2;

    ushort_t* xb     = (ushort_t*)alloc(MD2);
    ushort_t* wqt    = (ushort_t*)alloc(DD2);
    ushort_t* wkt    = (ushort_t*)alloc(DD2);
    ushort_t* wvt    = (ushort_t*)alloc(DD2);
    ushort_t* wot    = (ushort_t*)alloc(DD2);
    ushort_t* wff1t  = (ushort_t*)alloc(DF2);
    ushort_t* wff2t  = (ushort_t*)alloc(DF2);
    ushort_t* Qb     = (ushort_t*)alloc(MD2);
    ushort_t* Kb     = (ushort_t*)alloc(MD2);
    ushort_t* Vb     = (ushort_t*)alloc(MD2);
    ushort_t* ctxb   = (ushort_t*)alloc(MD2);
    float*    attnf  = (float*)alloc(MD4);
    float*    hf     = (float*)alloc(MD4);
    ushort_t* hb     = (ushort_t*)alloc(MD2);
    ushort_t* ff1b   = (ushort_t*)alloc(MF2);
    float*    fff    = (float*)alloc(MD4);

    // 1) convert inputs
    {
        int n = M_TOT * D_MODEL;
        cvt_f32_bf16<<<(n + 255) / 256, 256, 0, stream>>>(x, xb, n);
    }
    {
        int nDD = D_MODEL * D_MODEL;
        transpose_cvt<<<(nDD + 255) / 256, 256, 0, stream>>>(w_q, wqt, D_MODEL, D_MODEL);
        transpose_cvt<<<(nDD + 255) / 256, 256, 0, stream>>>(w_k, wkt, D_MODEL, D_MODEL);
        transpose_cvt<<<(nDD + 255) / 256, 256, 0, stream>>>(w_v, wvt, D_MODEL, D_MODEL);
        transpose_cvt<<<(nDD + 255) / 256, 256, 0, stream>>>(w_o, wot, D_MODEL, D_MODEL);
        int nDF = D_MODEL * D_FF;
        transpose_cvt<<<(nDF + 255) / 256, 256, 0, stream>>>(w_ff1, wff1t, D_MODEL, D_FF);
        transpose_cvt<<<(nDF + 255) / 256, 256, 0, stream>>>(w_ff2, wff2t, D_FF, D_MODEL);
    }

    // 2) QKV projections (bf16 out only)
    {
        dim3 grid(D_MODEL / 128, M_TOT / 128);
        gemm_bf16<<<grid, 256, 0, stream>>>(xb, wqt, b_q, nullptr, Qb, M_TOT, D_MODEL, D_MODEL, 0);
        gemm_bf16<<<grid, 256, 0, stream>>>(xb, wkt, b_k, nullptr, Kb, M_TOT, D_MODEL, D_MODEL, 0);
        gemm_bf16<<<grid, 256, 0, stream>>>(xb, wvt, b_v, nullptr, Vb, M_TOT, D_MODEL, D_MODEL, 0);
    }

    // 3) attention
    {
        dim3 grid(SEQ / 64, 2 * NHEADS);
        attn_kernel<<<grid, 128, 0, stream>>>(Qb, Kb, Vb, ctxb);
    }

    // 4) output projection (fp32 out)
    {
        dim3 grid(D_MODEL / 128, M_TOT / 128);
        gemm_bf16<<<grid, 256, 0, stream>>>(ctxb, wot, b_o, attnf, nullptr, M_TOT, D_MODEL, D_MODEL, 0);
    }

    // 5) AddNorm1: h = LN(x + attn_out)
    add_ln_kernel<<<M_TOT, 256, 0, stream>>>(x, attnf, ln1_g, ln1_b, hf, hb);

    // 6) FFN up + ReLU (bf16 out)
    {
        dim3 grid(D_FF / 128, M_TOT / 128);
        gemm_bf16<<<grid, 256, 0, stream>>>(hb, wff1t, b_ff1, nullptr, ff1b, M_TOT, D_FF, D_MODEL, 1);
    }

    // 7) FFN down (fp32 out)
    {
        dim3 grid(D_MODEL / 128, M_TOT / 128);
        gemm_bf16<<<grid, 256, 0, stream>>>(ff1b, wff2t, b_ff2, fff, nullptr, M_TOT, D_MODEL, D_FF, 0);
    }

    // 8) AddNorm2 -> output
    add_ln_kernel<<<M_TOT, 256, 0, stream>>>(hf, fff, ln2_g, ln2_b, out, nullptr);
}
